// CGCNNLayer_31559419691861
// MI455X (gfx1250) — matrix operand order repaired and checked by reading, not compiled, not run on hardware
//
#include <hip/hip_runtime.h>
#include <hip/hip_bf16.h>

// ---------- types ----------
typedef __bf16        bf16x16 __attribute__((ext_vector_type(16)));
typedef float         f32x8   __attribute__((ext_vector_type(8)));
typedef unsigned int  uintx4  __attribute__((ext_vector_type(4)));
typedef unsigned int  uintx2  __attribute__((ext_vector_type(2)));

union FragU {
    bf16x16 v;
    uintx4  q[2];
};

#define A_FEA     128
#define NBR_FEA   64
#define M_NBRS    12
#define IN_DIM    320          // 2*A + NBR_FEA
#define OUT_DIM   256          // 2*A
#define ATOMS_PB  4            // atoms per block
#define ROWS_PB   48           // ATOMS_PB * M_NBRS (3 row tiles of 16)
#define KSTEPS    10           // 320 / 32
#define ASTRIDE   328          // bf16 elements per LDS A row (320 + 8 pad)
#define CSTRIDE   260          // f32 elements per LDS C row (256 + 4 pad)
#define LN_EPS    1e-5f

// ---------- helpers ----------
__device__ __forceinline__ unsigned short f32_to_bf16(float f) {
    unsigned int u = __builtin_bit_cast(unsigned int, f);
    u += 0x7FFFu + ((u >> 16) & 1u);       // round-to-nearest-even
    return (unsigned short)(u >> 16);
}

__device__ __forceinline__ float softplusf(float x) {
    return fmaxf(x, 0.0f) + log1pf(expf(-fabsf(x)));
}
__device__ __forceinline__ float sigmoidf(float x) {
    return 1.0f / (1.0f + expf(-x));
}

// ---------- kernel 0: W (256x320 f32, row-major o,k) -> bf16 ----------
__global__ void cgcnn_wcvt(const float* __restrict__ W,
                           unsigned short* __restrict__ Wbf, int n) {
    int i = blockIdx.x * 256 + threadIdx.x;
    if (i < n) Wbf[i] = f32_to_bf16(W[i]);
}

// ---------- fused CGCNN layer kernel ----------
__global__ __launch_bounds__(256)
void cgcnn_fused(const float* __restrict__ atom_in,    // (N,128)
                 const float* __restrict__ nbr_atom,   // (N,12,128)
                 const float* __restrict__ nbr,        // (N,12,64)
                 const unsigned short* __restrict__ Wbf, // (256,320) bf16
                 const float* __restrict__ bvec,       // (256)
                 const float* __restrict__ gamma,      // (256)
                 const float* __restrict__ beta,       // (256)
                 float* __restrict__ out,              // (N,128)
                 int numAtoms) {
    __shared__ __align__(16) float smemC[ROWS_PB * CSTRIDE];   // 49920 B
    __shared__ float nbrSum[ATOMS_PB * A_FEA];                 // 2048 B
    unsigned short* smemA = (unsigned short*)smemC;            // aliased, 31488 B

    const int tid      = threadIdx.x;
    const int wave     = tid >> 5;
    const int lane     = tid & 31;
    const int laneLo   = lane & 15;
    const int hi       = lane >> 4;
    const int atomBase = blockIdx.x * ATOMS_PB;
    const int rowBase  = atomBase * M_NBRS;

    // zero the neighbor-sum accumulator
    nbrSum[tid]       = 0.0f;
    nbrSum[tid + 256] = 0.0f;

    // ---- stage A (concat + f32->bf16) into LDS: 48 rows x 320, float4 granules ----
    // per row: quads 0..31 from atom_in, 32..63 from nbr_atom, 64..79 from nbr
    for (int idx = tid; idx < ROWS_PB * 80; idx += 256) {
        int row = idx / 80;
        int q   = idx - row * 80;
        int nL  = row / M_NBRS;
        int gr  = rowBase + row;            // global (n*12+m)
        int n   = atomBase + nL;
        float4 val = make_float4(0.f, 0.f, 0.f, 0.f);
        if (n < numAtoms) {
            const float* src;
            if (q < 32)      src = atom_in  + ((long)n  * A_FEA   + q * 4);
            else if (q < 64) src = nbr_atom + ((long)gr * A_FEA   + (q - 32) * 4);
            else             src = nbr      + ((long)gr * NBR_FEA + (q - 64) * 4);
            val = *(const float4*)src;
        }
        uintx2 p;
        p.x = (unsigned int)f32_to_bf16(val.x) | ((unsigned int)f32_to_bf16(val.y) << 16);
        p.y = (unsigned int)f32_to_bf16(val.z) | ((unsigned int)f32_to_bf16(val.w) << 16);
        *(uintx2*)(smemA + row * ASTRIDE + q * 4) = p;
    }
    __syncthreads();

    // ---- WMMA GEMM: 3 row tiles x 2 col tiles per wave, K = 10 x 32 (bf16) ----
    f32x8 acc[3][2] = {};
    const int c0 = wave * 32 + laneLo;   // global output column for ct0
    const int c1 = c0 + 16;              // ct1

#pragma unroll 2
    for (int ks = 0; ks < KSTEPS; ++ks) {
        const int kb = ks * 32;
        FragU a[3], b0, b1;
#pragma unroll
        for (int r = 0; r < 3; ++r) {
            // A layout: lanes 0-15 hold K kb+0..15; lanes 16-31 hold K kb+16..31
            const unsigned short* ap = smemA + (r * 16 + laneLo) * ASTRIDE + kb + hi * 16;
            a[r].q[0] = *(const uintx4*)ap;
            a[r].q[1] = *(const uintx4*)(ap + 8);
        }
        // B layout: lanes 0-15 hold K kb+0..15; lanes 16-31 hold K kb+16..31 (col = lane%16)
        const unsigned short* bp0 = Wbf + c0 * IN_DIM + kb + hi * 16;
        b0.q[0] = *(const uintx4*)bp0;
        b0.q[1] = *(const uintx4*)(bp0 + 8);
        const unsigned short* bp1 = Wbf + c1 * IN_DIM + kb + hi * 16;
        b1.q[0] = *(const uintx4*)bp1;
        b1.q[1] = *(const uintx4*)(bp1 + 8);

#pragma unroll
        for (int r = 0; r < 3; ++r) {
            acc[r][0] = __builtin_amdgcn_wmma_f32_16x16x32_bf16(
                false, a[r].v, false, b0.v, (short)0, acc[r][0], false, false);
            acc[r][1] = __builtin_amdgcn_wmma_f32_16x16x32_bf16(
                false, a[r].v, false, b1.v, (short)0, acc[r][1], false, false);
        }
    }
    __syncthreads();   // A region dead; reuse LDS for C

    // ---- store C (+bias) to LDS ----
    {
        const float bias0 = bvec[c0];
        const float bias1 = bvec[c1];
#pragma unroll
        for (int r = 0; r < 3; ++r) {
#pragma unroll
            for (int v = 0; v < 8; ++v) {
                int row = r * 16 + v + 8 * hi;          // C layout: VGPR v -> M=v / v+8
                smemC[row * CSTRIDE + c0] = acc[r][0][v] + bias0;
                smemC[row * CSTRIDE + c1] = acc[r][1][v] + bias1;
            }
        }
    }
    __syncthreads();

    // ---- LayerNorm + gate + neighbor reduction (wave per row, 6 rows/wave) ----
#pragma unroll
    for (int rr = 0; rr < 6; ++rr) {
        int row = wave + rr * 8;
        int nL  = row / M_NBRS;
        if (atomBase + nL >= numAtoms) continue;        // uniform across wave
        const float* cr = smemC + row * CSTRIDE;

        float s = 0.f, s2 = 0.f;
#pragma unroll
        for (int t = 0; t < 8; ++t) {
            float x = cr[lane * 8 + t];
            s += x; s2 += x * x;
        }
        for (int off = 16; off > 0; off >>= 1) {
            s  += __shfl_xor(s,  off, 32);
            s2 += __shfl_xor(s2, off, 32);
        }
        float mean = s * (1.0f / OUT_DIM);
        float var  = s2 * (1.0f / OUT_DIM) - mean * mean;
        float inv  = rsqrtf(var + LN_EPS);

#pragma unroll
        for (int t = 0; t < 4; ++t) {
            int j = lane * 4 + t;
            float yf = (cr[j]       - mean) * inv * gamma[j]       + beta[j];
            float yc = (cr[j + 128] - mean) * inv * gamma[j + 128] + beta[j + 128];
            float g  = sigmoidf(yf) * softplusf(yc);
            atomicAdd(&nbrSum[nL * A_FEA + j], g);      // ds_add_f32
        }
    }
    __syncthreads();

    // ---- residual + softplus, write out ----
#pragma unroll
    for (int t = 0; t < 2; ++t) {
        int idx = tid + t * 256;
        int aL  = idx >> 7;
        int j   = idx & 127;
        int n   = atomBase + aL;
        if (n < numAtoms) {
            float v = atom_in[(long)n * A_FEA + j] + nbrSum[idx];
            out[(long)n * A_FEA + j] = softplusf(v);
        }
    }
}

extern "C" void kernel_launch(void* const* d_in, const int* in_sizes, int n_in,
                              void* d_out, int out_size, void* d_ws, size_t ws_size,
                              hipStream_t stream) {
    const float* atom_in  = (const float*)d_in[0];
    const float* nbr_atom = (const float*)d_in[1];
    const float* nbr      = (const float*)d_in[2];
    const float* W        = (const float*)d_in[3];
    const float* bvec     = (const float*)d_in[4];
    const float* gamma    = (const float*)d_in[5];
    const float* beta     = (const float*)d_in[6];
    float* out = (float*)d_out;

    const int numAtoms = in_sizes[0] / A_FEA;
    unsigned short* Wbf = (unsigned short*)d_ws;       // 256*320*2 = 160 KB scratch

    const int wTotal = OUT_DIM * IN_DIM;               // 81920
    cgcnn_wcvt<<<(wTotal + 255) / 256, 256, 0, stream>>>(W, Wbf, wTotal);

    const int blocks = (numAtoms + ATOMS_PB - 1) / ATOMS_PB;
    cgcnn_fused<<<blocks, 256, 0, stream>>>(atom_in, nbr_atom, nbr, Wbf,
                                            bvec, gamma, beta, out, numAtoms);
}
